// DesModelLoRA_20650202759789
// MI455X (gfx1250) — matrix operand order repaired and checked by reading, not compiled, not run on hardware
//
#include <hip/hip_runtime.h>
#include <hip/hip_bf16.h>

typedef __attribute__((ext_vector_type(16))) _Float16 v16h;
typedef __attribute__((ext_vector_type(8)))  _Float16 v8h;
typedef __attribute__((ext_vector_type(8)))  float    v8f;

#define BM 64
#define BN 64
#define BK 64
#define LDSW (BK + 8)   // 144B rows: every fragment read stays 16B aligned

enum { F_BIAS = 1, F_ACC = 2, F_OPBN = 4, F_GELU = 16, F_RELU = 32 };

// ---- stage one 64x64 fp32 tile pair into f16 LDS --------------------------
template<bool EDGE>
__device__ __forceinline__ void load_tiles(
    const float* __restrict__ A, long long lda,
    const float* __restrict__ W, long long ldw,
    int M, int N, int K, int bm0, int bn0, int k0, int tid, int flags,
    _Float16 (*As)[LDSW], _Float16 (*Bs)[LDSW])
{
  const int ar  = tid >> 2;          // 0..63
  const int acb = (tid & 3) * 4;     // 0..12
  #pragma unroll
  for (int cc = 0; cc < 4; cc++) {
    const int ac = acb + cc * 16;    // 0..60
    // A tile
    {
      float f0 = 0.f, f1 = 0.f, f2 = 0.f, f3 = 0.f;
      if (!EDGE) {
        float4 v = *(const float4*)(A + (long long)(bm0 + ar) * lda + k0 + ac);
        f0 = v.x; f1 = v.y; f2 = v.z; f3 = v.w;
      } else {
        const int grow = bm0 + ar;
        if (grow < M) {
          const float* p = A + (long long)grow * lda + k0 + ac;
          const int rem = K - (k0 + ac);
          if (rem >= 4) { float4 v = *(const float4*)p; f0 = v.x; f1 = v.y; f2 = v.z; f3 = v.w; }
          else { if (rem > 0) f0 = p[0]; if (rem > 1) f1 = p[1]; if (rem > 2) f2 = p[2]; }
        }
      }
      As[ar][ac + 0] = (_Float16)f0; As[ar][ac + 1] = (_Float16)f1;
      As[ar][ac + 2] = (_Float16)f2; As[ar][ac + 3] = (_Float16)f3;
    }
    // W tile, [N,K] layout
    if (!(flags & F_OPBN)) {
      float f0 = 0.f, f1 = 0.f, f2 = 0.f, f3 = 0.f;
      if (!EDGE) {
        float4 v = *(const float4*)(W + (long long)(bn0 + ar) * ldw + k0 + ac);
        f0 = v.x; f1 = v.y; f2 = v.z; f3 = v.w;
      } else {
        const int gcol = bn0 + ar;
        if (gcol < N) {
          const float* p = W + (long long)gcol * ldw + k0 + ac;
          const int rem = K - (k0 + ac);
          if (rem >= 4) { float4 v = *(const float4*)p; f0 = v.x; f1 = v.y; f2 = v.z; f3 = v.w; }
          else { if (rem > 0) f0 = p[0]; if (rem > 1) f1 = p[1]; if (rem > 2) f2 = p[2]; }
        }
      }
      Bs[ar][ac + 0] = (_Float16)f0; Bs[ar][ac + 1] = (_Float16)f1;
      Bs[ar][ac + 2] = (_Float16)f2; Bs[ar][ac + 3] = (_Float16)f3;
    }
  }
  // W tile, [K,N] layout (attn @ V)
  if (flags & F_OPBN) {
    const int nv  = (tid & 15) * 4;  // 0..60
    const int klb = tid >> 4;        // 0..15
    #pragma unroll
    for (int rr = 0; rr < 4; rr++) {
      const int kl = klb + rr * 16;
      const int k  = k0 + kl;
      float f0 = 0.f, f1 = 0.f, f2 = 0.f, f3 = 0.f;
      if (!EDGE) {
        float4 v = *(const float4*)(W + (long long)k * ldw + bn0 + nv);
        f0 = v.x; f1 = v.y; f2 = v.z; f3 = v.w;
      } else if (k < K) {
        const float* p = W + (long long)k * ldw + bn0 + nv;
        if (bn0 + nv + 3 < N) { float4 v = *(const float4*)p; f0 = v.x; f1 = v.y; f2 = v.z; f3 = v.w; }
        else {
          if (bn0 + nv + 0 < N) f0 = p[0];
          if (bn0 + nv + 1 < N) f1 = p[1];
          if (bn0 + nv + 2 < N) f2 = p[2];
        }
      }
      Bs[nv + 0][kl] = (_Float16)f0; Bs[nv + 1][kl] = (_Float16)f1;
      Bs[nv + 2][kl] = (_Float16)f2; Bs[nv + 3][kl] = (_Float16)f3;
    }
  }
}

// ---- 2 K-steps of WMMA per staged tile (4 x v_wmma_f32_16x16x32_f16) ------
__device__ __forceinline__ void mma_stage(
    const _Float16 (*As)[LDSW], const _Float16 (*Bs)[LDSW],
    int wm, int wn, int lane, v8f& acc0, v8f& acc1)
{
  const int hl  = lane >> 4;
  const int l16 = lane & 15;
  #pragma unroll
  for (int ks = 0; ks < 2; ks++) {
    const int kb = ks * 32;
    // A 16x32: lanes 0-15: K0-15 of row m ; lanes 16-31: K16-31
    v8h alo = *(const v8h*)&As[wm + l16][kb + hl * 16];
    v8h ahi = *(const v8h*)&As[wm + l16][kb + hl * 16 + 8];
    v16h af;
    #pragma unroll
    for (int i = 0; i < 8; i++) { af[i] = alo[i]; af[8 + i] = ahi[i]; }
    // B 32x16: lanes 0-15: K0-15 of col n ; lanes 16-31: K16-31
    v8h b0lo = *(const v8h*)&Bs[wn + l16][kb + hl * 16];
    v8h b0hi = *(const v8h*)&Bs[wn + l16][kb + hl * 16 + 8];
    v16h bf0;
    #pragma unroll
    for (int i = 0; i < 8; i++) { bf0[i] = b0lo[i]; bf0[8 + i] = b0hi[i]; }
    v8h b1lo = *(const v8h*)&Bs[wn + 16 + l16][kb + hl * 16];
    v8h b1hi = *(const v8h*)&Bs[wn + 16 + l16][kb + hl * 16 + 8];
    v16h bf1;
    #pragma unroll
    for (int i = 0; i < 8; i++) { bf1[i] = b1lo[i]; bf1[8 + i] = b1hi[i]; }

    acc0 = __builtin_amdgcn_wmma_f32_16x16x32_f16(false, af, false, bf0, (short)0, acc0, false, false);
    acc1 = __builtin_amdgcn_wmma_f32_16x16x32_f16(false, af, false, bf1, (short)0, acc1, false, false);
  }
}

// ---------------------------------------------------------------------------
// Generic batched GEMM: C = act(alpha * A @ op(W) + bias + (acc ? C : 0))
// Fast guard-free path for interior blocks; guarded path only on edges.
// ---------------------------------------------------------------------------
__global__ __launch_bounds__(256) void wmma_gemm_kernel(
    const float* __restrict__ Ab, long long lda, long long sAo, long long sAi,
    const float* __restrict__ Wb, long long ldw, long long sWo, long long sWi,
    const float* __restrict__ bias,
    float* __restrict__ Cb, long long ldc, long long sCo, long long sCi,
    int M, int N, int K, int binner, float alpha, int flags)
{
  __shared__ __align__(16) _Float16 As[BM][LDSW];
  __shared__ __align__(16) _Float16 Bs[BN][LDSW];

  const int z  = blockIdx.z;
  const int zo = z / binner, zi = z % binner;
  const float* A = Ab + zo * sAo + zi * sAi;
  const float* W = Wb + zo * sWo + zi * sWi;
  float*       C = Cb + zo * sCo + zi * sCi;

  const int bm0 = blockIdx.y * BM;
  const int bn0 = blockIdx.x * BN;
  const int tid  = threadIdx.x;
  const int lane = tid & 31;
  const int wv   = tid >> 5;
  const int wm   = (wv >> 1) * 16;   // 4 waves along M
  const int wn   = (wv & 1) * 32;    // 2 waves along N

  v8f acc0 = {0.f, 0.f, 0.f, 0.f, 0.f, 0.f, 0.f, 0.f};
  v8f acc1 = acc0;

  const bool interior = (bm0 + BM <= M) && (bn0 + BN <= N) && ((K & (BK - 1)) == 0);
  if (interior) {
    for (int k0 = 0; k0 < K; k0 += BK) {
      load_tiles<false>(A, lda, W, ldw, M, N, K, bm0, bn0, k0, tid, flags, As, Bs);
      if (k0 + BK < K)   // pull next A tile toward the caches (global_prefetch)
        __builtin_prefetch(A + (long long)(bm0 + (tid >> 2)) * lda + k0 + BK + (tid & 3) * 4, 0, 1);
      __syncthreads();
      mma_stage(As, Bs, wm, wn, lane, acc0, acc1);
      __syncthreads();
    }
  } else {
    for (int k0 = 0; k0 < K; k0 += BK) {
      load_tiles<true>(A, lda, W, ldw, M, N, K, bm0, bn0, k0, tid, flags, As, Bs);
      __syncthreads();
      mma_stage(As, Bs, wm, wn, lane, acc0, acc1);
      __syncthreads();
    }
  }

  // ---- epilogue: bias / accumulate / activation ----
  const int hl  = lane >> 4;
  const int l16 = lane & 15;
  const int act = (flags >> 4) & 3;
  #pragma unroll
  for (int sub = 0; sub < 2; sub++) {
    v8f acc = sub ? acc1 : acc0;
    const int col = bn0 + wn + sub * 16 + l16;
    #pragma unroll
    for (int r = 0; r < 8; r++) {
      const int row = bm0 + wm + r + hl * 8;
      if (row < M && col < N) {
        float v = alpha * acc[r];
        if (flags & F_BIAS) v += bias[col];
        if (flags & F_ACC)  v += C[(long long)row * ldc + col];
        if (act == 1)      v = 0.5f * v * (1.f + erff(v * 0.70710678118654752f)); // exact GELU
        else if (act == 2) v = fmaxf(v, 0.f);
        C[(long long)row * ldc + col] = v;
      }
    }
  }
}

// ---- wave-per-row LayerNorm (wave32 shuffle reduction) --------------------
__global__ void ln_kernel(const float* __restrict__ in, long long ipitch,
                          float* __restrict__ out, long long opitch,
                          const float* __restrict__ s, const float* __restrict__ b,
                          int nrows, int D_)
{
  const int row  = blockIdx.x * 8 + (threadIdx.x >> 5);
  const int lane = threadIdx.x & 31;
  if (row >= nrows) return;
  const float* x = in + (long long)row * ipitch;
  float sum = 0.f, sq = 0.f;
  for (int i = lane; i < D_; i += 32) { float v = x[i]; sum += v; sq += v * v; }
  for (int o = 16; o > 0; o >>= 1) { sum += __shfl_xor(sum, o, 32); sq += __shfl_xor(sq, o, 32); }
  const float inv  = 1.f / (float)D_;
  const float mean = sum * inv;
  const float var  = sq * inv - mean * mean;
  const float r    = rsqrtf(var + 1e-6f);
  float* y = out + (long long)row * opitch;
  for (int i = lane; i < D_; i += 32) y[i] = (x[i] - mean) * r * s[i] + b[i];
}

// ---- wave-per-row softmax (pitched, in place) -----------------------------
__global__ void softmax_kernel(float* __restrict__ p, long long pitch, int nrows, int n)
{
  const int row  = blockIdx.x * 8 + (threadIdx.x >> 5);
  const int lane = threadIdx.x & 31;
  if (row >= nrows) return;
  float* x = p + (long long)row * pitch;
  float mx = -3.402823466e38f;
  for (int i = lane; i < n; i += 32) mx = fmaxf(mx, x[i]);
  for (int o = 16; o > 0; o >>= 1) mx = fmaxf(mx, __shfl_xor(mx, o, 32));
  float sum = 0.f;
  for (int i = lane; i < n; i += 32) { float e = __expf(x[i] - mx); x[i] = e; sum += e; }
  for (int o = 16; o > 0; o >>= 1) sum += __shfl_xor(sum, o, 32);
  const float r = 1.f / sum;
  for (int i = lane; i < n; i += 32) x[i] *= r;
}

// img (16,3,384,384) -> patches (16*576, 768) with layout [c, i, j]
__global__ void gather_patches_kernel(const float* __restrict__ img,
                                      float* __restrict__ out, int total)
{
  const int idx = blockIdx.x * blockDim.x + threadIdx.x;
  if (idx >= total) return;
  const int c3 = idx % 768;
  const int t  = idx / 768;
  const int p  = t % 576;
  const int b  = t / 576;
  const int c  = c3 >> 8;
  const int ij = c3 & 255;
  const int i  = ij >> 4, j = ij & 15;
  const int ph = p / 24, pw = p % 24;
  out[idx] = img[(((long long)(b * 3 + c) * 384) + ph * 16 + i) * 384 + pw * 16 + j];
}

// x[b,0,:] = cls + pos[0]; x[b,n,:] = pe[b*576+n-1] + pos[n]
__global__ void assemble_x_kernel(const float* __restrict__ pe, const float* __restrict__ cls,
                                  const float* __restrict__ pos, float* __restrict__ x, int total)
{
  const int idx = blockIdx.x * blockDim.x + threadIdx.x;
  if (idx >= total) return;
  const int d = idx % 384;
  const int m = idx / 384;
  const int b = m / 577;
  const int n = m % 577;
  float v = pos[n * 384 + d];
  v += (n == 0) ? cls[d] : pe[(long long)(m - b - 1) * 384 + d];
  x[idx] = v;
}

static void launch_gemm(hipStream_t stream,
    const float* A, long long lda, long long sAo, long long sAi,
    const float* W, long long ldw, long long sWo, long long sWi,
    const float* bias, float* C, long long ldc, long long sCo, long long sCi,
    int M, int N, int K, int nb, int binner, float alpha, int flags)
{
  dim3 g((N + BN - 1) / BN, (M + BM - 1) / BM, nb);
  hipLaunchKernelGGL(wmma_gemm_kernel, g, dim3(256), 0, stream,
                     A, lda, sAo, sAi, W, ldw, sWo, sWi, bias,
                     C, ldc, sCo, sCi, M, N, K, binner, alpha, flags);
}

extern "C" void kernel_launch(void* const* d_in, const int* in_sizes, int n_in,
                              void* d_out, int out_size, void* d_ws, size_t ws_size,
                              hipStream_t stream)
{
  const float* img     = (const float*)d_in[0];
  const float* patch_w = (const float*)d_in[1];
  const float* patch_b = (const float*)d_in[2];
  const float* cls_tok = (const float*)d_in[3];
  const float* pos     = (const float*)d_in[4];
  const float* ln1_s   = (const float*)d_in[5];
  const float* ln1_b   = (const float*)d_in[6];
  const float* qkv_w   = (const float*)d_in[7];
  const float* qkv_b   = (const float*)d_in[8];
  const float* qkv_A   = (const float*)d_in[9];
  const float* qkv_B   = (const float*)d_in[10];
  const float* proj_w  = (const float*)d_in[11];
  const float* proj_b  = (const float*)d_in[12];
  const float* proj_A  = (const float*)d_in[13];
  const float* proj_B  = (const float*)d_in[14];
  const float* ln2_s   = (const float*)d_in[15];
  const float* ln2_b   = (const float*)d_in[16];
  const float* fc1_w   = (const float*)d_in[17];
  const float* fc1_b   = (const float*)d_in[18];
  const float* fc1_A   = (const float*)d_in[19];
  const float* fc1_B   = (const float*)d_in[20];
  const float* fc2_w   = (const float*)d_in[21];
  const float* fc2_b   = (const float*)d_in[22];
  const float* fc2_A   = (const float*)d_in[23];
  const float* fc2_B   = (const float*)d_in[24];
  const float* norm_s  = (const float*)d_in[25];
  const float* norm_b  = (const float*)d_in[26];
  const float* head_w1 = (const float*)d_in[27];
  const float* head_b1 = (const float*)d_in[28];
  const float* head_w2 = (const float*)d_in[29];
  const float* head_b2 = (const float*)d_in[30];
  float* out = (float*)d_out;

  const int Bb = 16, D = 384, R = 128, NH = 6, HD = 64, MLPD = 1536, NT = 577, L = 12;
  const int M = Bb * NT;                 // 9232
  const int LDS_SC = 580;                // padded scores leading dim (16B rows)
  const float ATTN_SCALE = 0.125f;       // 1/sqrt(64)

  float* ws = (float*)d_ws;
  size_t o_x   = 0;
  size_t o_h   = o_x   + (size_t)M * D;
  size_t o_t   = o_h   + (size_t)M * D;
  size_t o_qkv = o_t   + (size_t)M * R;
  size_t o_o   = o_qkv + (size_t)M * 3 * D;
  size_t o_a   = o_o   + (size_t)M * D;
  size_t o_sc  = o_a   + (size_t)M * MLPD;
  size_t o_cls = o_sc  + (size_t)Bb * NH * NT * LDS_SC;
  size_t o_h1  = o_cls + (size_t)Bb * D;

  float* X   = ws + o_x;
  float* H   = ws + o_h;
  float* T   = ws + o_t;
  float* QKV = ws + o_qkv;
  float* O   = ws + o_o;
  float* AA  = ws + o_a;
  float* SC  = ws + o_sc;
  float* CLS = ws + o_cls;
  float* H1  = ws + o_h1;
  // patch-embed staging aliases the (not-yet-used) scores region
  float* PT = SC;                          // [9216, 768]
  float* PE = SC + (size_t)9216 * 768;     // [9216, 384]

  // ---- patch embed + cls + pos ----
  {
    const int tot_p = Bb * 576 * 768;
    gather_patches_kernel<<<(tot_p + 255) / 256, 256, 0, stream>>>(img, PT, tot_p);
    launch_gemm(stream, PT, 768, 0, 0, patch_w, 768, 0, 0, patch_b,
                PE, D, 0, 0, Bb * 576, D, 768, 1, 1, 1.f, F_BIAS);
    const int tot_x = M * D;
    assemble_x_kernel<<<(tot_x + 255) / 256, 256, 0, stream>>>(PE, cls_tok, pos, X, tot_x);
  }

  // ---- transformer blocks ----
  for (int l = 0; l < L; l++) {
    const float* l1s = ln1_s + (size_t)l * D;
    const float* l1b = ln1_b + (size_t)l * D;
    const float* qw  = qkv_w  + (size_t)l * 3 * D * D;
    const float* qb  = qkv_b  + (size_t)l * 3 * D;
    const float* qA  = qkv_A  + (size_t)l * R * D;
    const float* qB  = qkv_B  + (size_t)l * 3 * D * R;
    const float* pw  = proj_w + (size_t)l * D * D;
    const float* pb  = proj_b + (size_t)l * D;
    const float* pA  = proj_A + (size_t)l * R * D;
    const float* pB  = proj_B + (size_t)l * D * R;
    const float* l2s = ln2_s + (size_t)l * D;
    const float* l2b = ln2_b + (size_t)l * D;
    const float* f1w = fc1_w + (size_t)l * MLPD * D;
    const float* f1b = fc1_b + (size_t)l * MLPD;
    const float* f1A = fc1_A + (size_t)l * R * D;
    const float* f1B = fc1_B + (size_t)l * MLPD * R;
    const float* f2w = fc2_w + (size_t)l * D * MLPD;
    const float* f2b = fc2_b + (size_t)l * D;
    const float* f2A = fc2_A + (size_t)l * R * MLPD;
    const float* f2B = fc2_B + (size_t)l * D * R;

    // h = LN1(x)
    ln_kernel<<<(M + 7) / 8, 256, 0, stream>>>(X, D, H, D, l1s, l1b, M, D);
    // qkv = h@qw^T + qb + 2*(h@qA^T)@qB^T
    launch_gemm(stream, H, D, 0, 0, qA, D, 0, 0, nullptr, T, R, 0, 0, M, R, D, 1, 1, 1.f, 0);
    launch_gemm(stream, H, D, 0, 0, qw, D, 0, 0, qb, QKV, 3 * D, 0, 0, M, 3 * D, D, 1, 1, 1.f, F_BIAS);
    launch_gemm(stream, T, R, 0, 0, qB, R, 0, 0, nullptr, QKV, 3 * D, 0, 0, M, 3 * D, R, 1, 1, 2.f, F_ACC);
    // scores[b,h] = scale * Q@K^T  (strided directly over QKV buffer)
    launch_gemm(stream,
                QKV,      3 * D, (long long)NT * 3 * D, HD,
                QKV + D,  3 * D, (long long)NT * 3 * D, HD,
                nullptr,
                SC, LDS_SC, (long long)NH * NT * LDS_SC, (long long)NT * LDS_SC,
                NT, NT, HD, Bb * NH, NH, ATTN_SCALE, 0);
    softmax_kernel<<<(Bb * NH * NT + 7) / 8, 256, 0, stream>>>(SC, LDS_SC, Bb * NH * NT, NT);
    // O[b,n,h,d] = attn @ V   (V accessed as [K,N] via F_OPBN)
    launch_gemm(stream,
                SC, LDS_SC, (long long)NH * NT * LDS_SC, (long long)NT * LDS_SC,
                QKV + 2 * D, 3 * D, (long long)NT * 3 * D, HD,
                nullptr,
                O, D, (long long)NT * D, HD,
                NT, HD, NT, Bb * NH, NH, 1.f, F_OPBN);
    // x += O@pw^T + pb + 2*(O@pA^T)@pB^T
    launch_gemm(stream, O, D, 0, 0, pA, D, 0, 0, nullptr, T, R, 0, 0, M, R, D, 1, 1, 1.f, 0);
    launch_gemm(stream, O, D, 0, 0, pw, D, 0, 0, pb, X, D, 0, 0, M, D, D, 1, 1, 1.f, F_BIAS | F_ACC);
    launch_gemm(stream, T, R, 0, 0, pB, R, 0, 0, nullptr, X, D, 0, 0, M, D, R, 1, 1, 2.f, F_ACC);
    // MLP
    ln_kernel<<<(M + 7) / 8, 256, 0, stream>>>(X, D, H, D, l2s, l2b, M, D);
    launch_gemm(stream, H, D, 0, 0, f1A, D, 0, 0, nullptr, T, R, 0, 0, M, R, D, 1, 1, 1.f, 0);
    launch_gemm(stream, H, D, 0, 0, f1w, D, 0, 0, f1b, AA, MLPD, 0, 0, M, MLPD, D, 1, 1, 1.f, F_BIAS);
    launch_gemm(stream, T, R, 0, 0, f1B, R, 0, 0, nullptr, AA, MLPD, 0, 0, M, MLPD, R, 1, 1, 2.f, F_ACC | F_GELU);
    launch_gemm(stream, AA, MLPD, 0, 0, f2A, MLPD, 0, 0, nullptr, T, R, 0, 0, M, R, MLPD, 1, 1, 1.f, 0);
    launch_gemm(stream, AA, MLPD, 0, 0, f2w, MLPD, 0, 0, f2b, X, D, 0, 0, M, D, MLPD, 1, 1, 1.f, F_BIAS | F_ACC);
    launch_gemm(stream, T, R, 0, 0, f2B, R, 0, 0, nullptr, X, D, 0, 0, M, D, R, 1, 1, 2.f, F_ACC);
  }

  // ---- final LN on CLS rows only + head MLP ----
  ln_kernel<<<(Bb + 7) / 8, 256, 0, stream>>>(X, (long long)NT * D, CLS, D, norm_s, norm_b, Bb, D);
  launch_gemm(stream, CLS, D, 0, 0, head_w1, D, 0, 0, head_b1,
              H1, 2048, 0, 0, Bb, 2048, D, 1, 1, 1.f, F_BIAS | F_RELU);
  launch_gemm(stream, H1, 2048, 0, 0, head_w2, 2048, 0, 0, head_b2,
              out, 256, 0, 0, Bb, 256, 2048, 1, 1, 1.f, F_BIAS);
}
